// EnergyForceModel_44873818309208
// MI455X (gfx1250) — matrix operand order repaired and checked by reading, not compiled, not run on hardware
//
#include <hip/hip_runtime.h>
#include <hip/hip_bf16.h>

#define NN      1024
#define DD      22
#define DD2     44
#define HH      64
#define NF      (NN*DD)      // 22528
#define EPS_TRI 1e-5f
#define EPS_LN  1e-6f
#define LDSTR   40           // padded LDS row stride (bf16 elems): 80B -> conflict-free

typedef __attribute__((ext_vector_type(16))) __bf16 v16bf;
typedef __attribute__((ext_vector_type(8)))  __bf16 v8bf;
typedef __attribute__((ext_vector_type(8)))  float  v8f;

__device__ __forceinline__ float sigmoidf_(float x) { return 1.0f / (1.0f + __expf(-x)); }
__device__ __forceinline__ float siluf_(float x)    { return x * sigmoidf_(x); }

// ---------------------------------------------------------------------------
// Kernel 1: pair features -> layernorm -> a/b projections (bf16) + out-gate
// A[d][i][k], B[d][j][k]  (k == second pair index), G[i][j][d]
// ---------------------------------------------------------------------------
__global__ void __launch_bounds__(256) k_pair_build(
    const float* __restrict__ pos, const int* __restrict__ anum,
    const float* __restrict__ q,
    const float* __restrict__ niw, const float* __restrict__ nib,
    const float* __restrict__ piw, const float* __restrict__ pib,
    const float* __restrict__ giw, const float* __restrict__ gib,
    const float* __restrict__ gow, const float* __restrict__ gob,
    __hip_bfloat16* __restrict__ Abf,
    __hip_bfloat16* __restrict__ Bbf,
    __hip_bfloat16* __restrict__ Gbf)
{
    __shared__ float s_piw[DD2*DD], s_giw[DD2*DD], s_gow[DD*DD];
    __shared__ float s_pib[DD2], s_gib[DD2], s_gob[DD], s_niw[DD], s_nib[DD];
    for (int t = threadIdx.x; t < DD2*DD; t += 256) { s_piw[t] = piw[t]; s_giw[t] = giw[t]; }
    for (int t = threadIdx.x; t < DD*DD;  t += 256) { s_gow[t] = gow[t]; }
    if (threadIdx.x < DD2) { s_pib[threadIdx.x] = pib[threadIdx.x]; s_gib[threadIdx.x] = gib[threadIdx.x]; }
    if (threadIdx.x < DD)  { s_gob[threadIdx.x] = gob[threadIdx.x];
                             s_niw[threadIdx.x] = niw[threadIdx.x];
                             s_nib[threadIdx.x] = nib[threadIdx.x]; }
    __syncthreads();

    const int idx = blockIdx.x * 256 + threadIdx.x;     // exact: N*N / 256 blocks
    const int i = idx >> 10;
    const int j = idx & (NN - 1);

    const float dx = pos[3*i+0] - pos[3*j+0];
    const float dy = pos[3*i+1] - pos[3*j+1];
    const float dz = pos[3*i+2] - pos[3*j+2];
    const float ex = dx + 1e-9f, ey = dy + 1e-9f, ez = dz + 1e-9f;
    const float dist = sqrtf(ex*ex + ey*ey + ez*ez);
    const float inv  = 1.0f / (dist + 1e-9f);
    const float ux = dx*inv, uy = dy*inv, uz = dz*inv;
    const float r2 = ux*ux + uy*uy + uz*uz;
    const float s3 = 1.7320508075688772f, s5 = 2.2360679774997896f, s15 = 3.872983346207417f;

    float x[DD];
    x[0] = dist;
    float sh[9];
    sh[0] = 1.0f;            sh[1] = s3*ux;          sh[2] = s3*uy;  sh[3] = s3*uz;
    sh[4] = s15*ux*uy;       sh[5] = s15*uy*uz;      sh[6] = 0.5f*s5*(3.0f*uz*uz - r2);
    sh[7] = s15*uz*ux;       sh[8] = 0.5f*s15*(ux*ux - uy*uy);
    #pragma unroll
    for (int k = 0; k < 9; ++k) { x[1+k] = sh[k]; x[10+k] = sh[k]; }
    x[19] = (float)anum[i];
    x[20] = (float)anum[j];
    x[21] = q[0];

    // layernorm over 22
    float m = 0.f, sq = 0.f;
    #pragma unroll
    for (int d = 0; d < DD; ++d) { m += x[d]; sq += x[d]*x[d]; }
    m *= (1.0f/DD); sq *= (1.0f/DD);
    const float rstd = rsqrtf(sq - m*m + EPS_TRI);
    float xn[DD];
    #pragma unroll
    for (int d = 0; d < DD; ++d) xn[d] = (x[d]-m)*rstd*s_niw[d] + s_nib[d];

    // ab = (xn @ piw.T + pib) * sigmoid(xn @ giw.T + gib)
    #pragma unroll 4
    for (int o = 0; o < DD2; ++o) {
        float p = s_pib[o], g = s_gib[o];
        #pragma unroll
        for (int d = 0; d < DD; ++d) { p += xn[d]*s_piw[o*DD+d]; g += xn[d]*s_giw[o*DD+d]; }
        const float ab = p * sigmoidf_(g);
        if (o < DD) Abf[(size_t)o*NN*NN + (size_t)i*NN + j] = __float2bfloat16(ab);
        else        Bbf[(size_t)(o-DD)*NN*NN + (size_t)i*NN + j] = __float2bfloat16(ab);
    }
    // out gate = sigmoid(xn @ gow.T + gob)
    #pragma unroll 2
    for (int o = 0; o < DD; ++o) {
        float g = s_gob[o];
        #pragma unroll
        for (int d = 0; d < DD; ++d) g += xn[d]*s_gow[o*DD+d];
        Gbf[(size_t)idx*DD + o] = __float2bfloat16(sigmoidf_(g));
    }
}

// ---------------------------------------------------------------------------
// Kernel 2: triangle einsum — 22 channel GEMMs T_d = A_d * B_d^T, bf16 WMMA.
// Block = 8 waves (4x2), block tile 128x64, wave tile 32x32 (2x2 WMMA tiles).
// Double-buffered LDS staging: A/B tiles fetched once per block per K-step.
// ---------------------------------------------------------------------------
__global__ void __launch_bounds__(256) k_triangle_wmma(
    const __hip_bfloat16* __restrict__ Abf,
    const __hip_bfloat16* __restrict__ Bbf,
    float* __restrict__ T)
{
    __shared__ __align__(16) __bf16 sA[2][128*LDSTR];   // 2 x 10 KB
    __shared__ __align__(16) __bf16 sB[2][64*LDSTR];    // 2 x  5 KB

    const int d    = blockIdx.y;
    const int t    = threadIdx.x;
    const int lane = t & 31;
    const int wave = t >> 5;
    const int wm   = wave & 3;          // 4 row sub-tiles (32 rows each)
    const int wn   = wave >> 2;         // 2 col sub-tiles (32 cols each)
    const int bi   = (blockIdx.x & (NN/128 - 1)) * 128;
    const int bj   = (blockIdx.x / (NN/128)) * 64;
    const int lm   = lane & 15;         // row/col within 16x16 tile
    const int lh   = lane >> 4;         // K-half select

    const __hip_bfloat16* __restrict__ Ag = Abf + (size_t)d*NN*NN;
    const __hip_bfloat16* __restrict__ Bg = Bbf + (size_t)d*NN*NN;

    // Cooperative tile fetch mapping: thread -> (row = t/4, 16B chunk = t%4)
    const int cr = t >> 2;              // 0..63
    const int cs = (t & 3) * 8;         // k-chunk offset in elements
    const __hip_bfloat16* __restrict__ gA0 = Ag + (size_t)(bi + cr)*NN + cs;       // rows 0..63
    const __hip_bfloat16* __restrict__ gA1 = gA0 + (size_t)64*NN;                  // rows 64..127
    const __hip_bfloat16* __restrict__ gB  = Bg + (size_t)(bj + cr)*NN + cs;       // rows 0..63
    const int la0 = cr*LDSTR + cs;
    const int la1 = la0 + 64*LDSTR;
    const int lb  = cr*LDSTR + cs;

    // Prologue: stage K-step 0 into buffer 0
    {
        const uint4 ra0 = *(const uint4*)gA0;
        const uint4 ra1 = *(const uint4*)gA1;
        const uint4 rb  = *(const uint4*)gB;
        *(uint4*)&sA[0][la0] = ra0;
        *(uint4*)&sA[0][la1] = ra1;
        *(uint4*)&sB[0][lb]  = rb;
    }

    v8f c00 = {}, c01 = {}, c10 = {}, c11 = {};

    #pragma unroll 4
    for (int s = 0; s < NN/32; ++s) {
        const int k0 = s*32;
        uint4 na0, na1, nb;
        const bool more = (s + 1) < NN/32;
        if (more) {                     // issue next-step global loads early
            na0 = *(const uint4*)(gA0 + k0 + 32);
            na1 = *(const uint4*)(gA1 + k0 + 32);
            nb  = *(const uint4*)(gB  + k0 + 32);
        }
        __syncthreads();                // current buffer visible to all waves

        const __bf16* __restrict__ pa = &sA[s & 1][(wm*32 + lm)*LDSTR];
        const __bf16* __restrict__ pb = &sB[s & 1][(wn*32 + lm)*LDSTR];

        // A fragments: elems 0..7 = K[lh*8..], 8..15 = K[16+lh*8..]
        const v8bf a00 = *(const v8bf*)(pa + lh*8);
        const v8bf a01 = *(const v8bf*)(pa + 16 + lh*8);
        const v8bf a10 = *(const v8bf*)(pa + 16*LDSTR + lh*8);
        const v8bf a11 = *(const v8bf*)(pa + 16*LDSTR + 16 + lh*8);
        const v16bf av0 = __builtin_shufflevector(a00, a01,
            0,1,2,3,4,5,6,7,8,9,10,11,12,13,14,15);
        const v16bf av1 = __builtin_shufflevector(a10, a11,
            0,1,2,3,4,5,6,7,8,9,10,11,12,13,14,15);
        // B fragments: lane lm col; elems 0..7 = K[lh*8..], 8..15 = K[16+lh*8..]
        const v8bf b00 = *(const v8bf*)(pb + lh*8);
        const v8bf b01 = *(const v8bf*)(pb + 16 + lh*8);
        const v8bf b10 = *(const v8bf*)(pb + 16*LDSTR + lh*8);
        const v8bf b11 = *(const v8bf*)(pb + 16*LDSTR + 16 + lh*8);
        const v16bf bv0 = __builtin_shufflevector(b00, b01,
            0,1,2,3,4,5,6,7,8,9,10,11,12,13,14,15);
        const v16bf bv1 = __builtin_shufflevector(b10, b11,
            0,1,2,3,4,5,6,7,8,9,10,11,12,13,14,15);

        c00 = __builtin_amdgcn_wmma_f32_16x16x32_bf16(false, av0, false, bv0, (short)0, c00, false, false);
        c01 = __builtin_amdgcn_wmma_f32_16x16x32_bf16(false, av0, false, bv1, (short)0, c01, false, false);
        c10 = __builtin_amdgcn_wmma_f32_16x16x32_bf16(false, av1, false, bv0, (short)0, c10, false, false);
        c11 = __builtin_amdgcn_wmma_f32_16x16x32_bf16(false, av1, false, bv1, (short)0, c11, false, false);

        if (more) {                     // fill alternate buffer (safe: last read s-1)
            *(uint4*)&sA[(s+1) & 1][la0] = na0;
            *(uint4*)&sA[(s+1) & 1][la1] = na1;
            *(uint4*)&sB[(s+1) & 1][lb]  = nb;
        }
    }

    // C/D layout: VGPR r -> (M = r + 8*lh, N = lm)
    float* __restrict__ Td = T + (size_t)d*NN*NN;
    const int ti   = bi + wm*32;
    const int tj   = bj + wn*32;
    const int col0 = tj + lm;
    const int row0 = ti + lh*8;
    #pragma unroll
    for (int r = 0; r < 8; ++r) {
        Td[(size_t)(row0 + r)*NN      + col0]      = c00[r];
        Td[(size_t)(row0 + r)*NN      + col0 + 16] = c01[r];
        Td[(size_t)(row0 + 16 + r)*NN + col0]      = c10[r];
        Td[(size_t)(row0 + 16 + r)*NN + col0 + 16] = c11[r];
    }
}

// ---------------------------------------------------------------------------
// Kernel 3: layernorm(t) @ pow.T + pow_b, times stored gate -> pair output
// PAIRO[i][j*22 + o]  (== atom_features row-major reshape)
// ---------------------------------------------------------------------------
__global__ void __launch_bounds__(256) k_tri_out(
    const float* __restrict__ T,
    const __hip_bfloat16* __restrict__ Gbf,
    const float* __restrict__ now, const float* __restrict__ nob,
    const float* __restrict__ poww, const float* __restrict__ powb,
    float* __restrict__ PAIRO)
{
    __shared__ float s_pow[DD*DD], s_powb[DD], s_now[DD], s_nob[DD];
    for (int t = threadIdx.x; t < DD*DD; t += 256) s_pow[t] = poww[t];
    if (threadIdx.x < DD) { s_powb[threadIdx.x] = powb[threadIdx.x];
                            s_now[threadIdx.x]  = now[threadIdx.x];
                            s_nob[threadIdx.x]  = nob[threadIdx.x]; }
    __syncthreads();

    const size_t idx = (size_t)blockIdx.x * 256 + threadIdx.x;

    float t[DD];
    #pragma unroll
    for (int d = 0; d < DD; ++d) t[d] = T[(size_t)d*NN*NN + idx];

    float m = 0.f, sq = 0.f;
    #pragma unroll
    for (int d = 0; d < DD; ++d) { m += t[d]; sq += t[d]*t[d]; }
    m *= (1.0f/DD); sq *= (1.0f/DD);
    const float rstd = rsqrtf(sq - m*m + EPS_TRI);
    float tn[DD];
    #pragma unroll
    for (int d = 0; d < DD; ++d) tn[d] = (t[d]-m)*rstd*s_now[d] + s_nob[d];

    #pragma unroll 2
    for (int o = 0; o < DD; ++o) {
        float p = s_powb[o];
        #pragma unroll
        for (int d = 0; d < DD; ++d) p += tn[d]*s_pow[o*DD+d];
        const float g = __bfloat162float(Gbf[idx*DD + o]);
        PAIRO[idx*DD + o] = p * g;   // idx*22+o == i*NF + j*22 + o
    }
}

// ---------------------------------------------------------------------------
// Kernel 4: per-atom head. One block per atom.
// layernorm(22528) -> W1(22528x64)+silu -> W2 -> W3 -> Wo -> per_atom[i]
// ---------------------------------------------------------------------------
__global__ void __launch_bounds__(256) k_atom_head(
    const float* __restrict__ PAIRO,
    const float* __restrict__ lns, const float* __restrict__ lnb,
    const float* __restrict__ W1, const float* __restrict__ b1,
    const float* __restrict__ W2, const float* __restrict__ b2,
    const float* __restrict__ W3, const float* __restrict__ b3,
    const float* __restrict__ Wo, const float* __restrict__ bo,
    const float* __restrict__ mask,
    float* __restrict__ per_atom)
{
    __shared__ float red[256];
    __shared__ float red2[256];
    __shared__ float parts[4*HH];
    __shared__ float sh1[HH], sh2[HH], sh3[HH];

    const int i = blockIdx.x;
    const float* __restrict__ xrow = PAIRO + (size_t)i * NF;
    const int t = threadIdx.x;

    // mean / var over 22528
    float s = 0.f, sq = 0.f;
    for (int f = t; f < NF; f += 256) { const float v = xrow[f]; s += v; sq += v*v; }
    red[t] = s; red2[t] = sq; __syncthreads();
    for (int off = 128; off > 0; off >>= 1) {
        if (t < off) { red[t] += red[t+off]; red2[t] += red2[t+off]; }
        __syncthreads();
    }
    const float m    = red[0] * (1.0f/NF);
    const float var  = red2[0] * (1.0f/NF) - m*m;
    const float rstd = rsqrtf(var + EPS_LN);

    // h1 = silu(LN(x) @ W1 + b1): 4 groups x 64 lanes
    const int h = t & (HH-1);
    const int g = t >> 6;
    float acc = 0.f;
    const int f0 = g * (NF/4), f1 = f0 + (NF/4);
    for (int f = f0; f < f1; ++f) {
        const float xl = (xrow[f]-m)*rstd*lns[f] + lnb[f];
        acc += xl * W1[(size_t)f*HH + h];
    }
    parts[g*HH + h] = acc; __syncthreads();
    if (t < HH) {
        const float v = parts[t] + parts[HH+t] + parts[2*HH+t] + parts[3*HH+t] + b1[t];
        sh1[t] = siluf_(v);
    }
    __syncthreads();
    if (t < HH) {
        float v = b2[t];
        #pragma unroll 8
        for (int k = 0; k < HH; ++k) v += sh1[k]*W2[k*HH+t];
        sh2[t] = siluf_(v);
    }
    __syncthreads();
    if (t < HH) {
        float v = b3[t];
        #pragma unroll 8
        for (int k = 0; k < HH; ++k) v += sh2[k]*W3[k*HH+t];
        sh3[t] = siluf_(v);
    }
    __syncthreads();
    if (t == 0) {
        float e = bo[0];
        #pragma unroll 8
        for (int k = 0; k < HH; ++k) e += sh3[k]*Wo[k];
        per_atom[i] = e * mask[i];
    }
}

// Final deterministic tree reduction of 1024 per-atom energies -> d_out[0]
__global__ void __launch_bounds__(256) k_reduce(const float* __restrict__ per_atom,
                                                float* __restrict__ out)
{
    __shared__ float red[256];
    const int t = threadIdx.x;
    float s = per_atom[t] + per_atom[t+256] + per_atom[t+512] + per_atom[t+768];
    red[t] = s; __syncthreads();
    for (int off = 128; off > 0; off >>= 1) {
        if (t < off) red[t] += red[t+off];
        __syncthreads();
    }
    if (t == 0) out[0] = red[0];
}

// ---------------------------------------------------------------------------
extern "C" void kernel_launch(void* const* d_in, const int* in_sizes, int n_in,
                              void* d_out, int out_size, void* d_ws, size_t ws_size,
                              hipStream_t stream) {
    const float* pos  = (const float*)d_in[0];
    const int*   anum = (const int*)  d_in[1];
    const float* q    = (const float*)d_in[2];
    const float* mask = (const float*)d_in[3];
    const float* niw  = (const float*)d_in[4];
    const float* nib  = (const float*)d_in[5];
    const float* piw  = (const float*)d_in[6];
    const float* pib  = (const float*)d_in[7];
    const float* giw  = (const float*)d_in[8];
    const float* gib  = (const float*)d_in[9];
    const float* now  = (const float*)d_in[10];
    const float* nob  = (const float*)d_in[11];
    const float* poww = (const float*)d_in[12];
    const float* powb = (const float*)d_in[13];
    const float* gow  = (const float*)d_in[14];
    const float* gob  = (const float*)d_in[15];
    const float* lns  = (const float*)d_in[16];
    const float* lnb  = (const float*)d_in[17];
    const float* W1   = (const float*)d_in[18];
    const float* b1   = (const float*)d_in[19];
    const float* W2   = (const float*)d_in[20];
    const float* b2   = (const float*)d_in[21];
    const float* W3   = (const float*)d_in[22];
    const float* b3   = (const float*)d_in[23];
    const float* Wo   = (const float*)d_in[24];
    const float* bo   = (const float*)d_in[25];

    char* base = (char*)d_ws;
    const size_t szAB   = (size_t)DD*NN*NN*sizeof(__hip_bfloat16);   // 46.1 MB
    const size_t szG    = (size_t)NN*NN*DD*sizeof(__hip_bfloat16);   // 46.1 MB
    const size_t szT    = (size_t)DD*NN*NN*sizeof(float);            // 92.3 MB
    const size_t szP    = (size_t)NN*NF*sizeof(float);               // 92.3 MB
    __hip_bfloat16* Abf = (__hip_bfloat16*)(base);
    __hip_bfloat16* Bbf = (__hip_bfloat16*)(base + szAB);
    __hip_bfloat16* Gbf = (__hip_bfloat16*)(base + 2*szAB);
    float*          T   = (float*)(base + 2*szAB + szG);
    float*          P   = (float*)(base + 2*szAB + szG + szT);
    float*          pa  = (float*)(base + 2*szAB + szG + szT + szP);
    float*          out = (float*)d_out;

    // 1) pair features + a/b/gate
    k_pair_build<<<(NN*NN)/256, 256, 0, stream>>>(pos, anum, q, niw, nib,
        piw, pib, giw, gib, gow, gob, Abf, Bbf, Gbf);

    // 2) 22 channel GEMMs via bf16 WMMA (LDS double-buffered, 2x2 blocked waves)
    dim3 grid2((NN/128)*(NN/64), DD);
    k_triangle_wmma<<<grid2, 256, 0, stream>>>(Abf, Bbf, T);

    // 3) output layernorm + projection + gate
    k_tri_out<<<(NN*NN)/256, 256, 0, stream>>>(T, Gbf, now, nob, poww, powb, P);

    // 4) per-atom MLP head
    k_atom_head<<<NN, 256, 0, stream>>>(P, lns, lnb, W1, b1, W2, b2, W3, b3,
                                        Wo, bo, mask, pa);

    // 5) deterministic final sum
    k_reduce<<<1, 256, 0, stream>>>(pa, out);
}
